// align_uniform_Loss_6176162971828
// MI455X (gfx1250) — matrix operand order repaired and checked by reading, not compiled, not run on hardware
//
#include <hip/hip_runtime.h>
#include <hip/hip_bf16.h>
#include <math.h>

typedef __attribute__((ext_vector_type(16))) _Float16 v16h;
typedef __attribute__((ext_vector_type(8)))  _Float16 v8h;
typedef __attribute__((ext_vector_type(4)))  _Float16 v4h;
typedef __attribute__((ext_vector_type(8)))  float    v8f;

#define N_ROWS 8192
#define DIM    128

__device__ __forceinline__ float wave_sum(float v) {
#pragma unroll
    for (int o = 16; o > 0; o >>= 1) v += __shfl_xor(v, o, 32);
    return v;
}

// acc[0] = sum_i ||u_i - p_i||^2 ; acc[1] = S_full(users) ; acc[2] = S_full(items)
__global__ void au_init_acc(float* acc) {
    if (threadIdx.x < 4) acc[threadIdx.x] = 0.0f;
}

// One wave (32 lanes) per row. Each lane holds 4 floats (128/32).
__global__ void au_normalize(const float* __restrict__ users,
                             const float* __restrict__ items,
                             _Float16* __restrict__ Xu,
                             _Float16* __restrict__ Xi,
                             float* __restrict__ acc) {
    const int wave = threadIdx.x >> 5;
    const int lane = threadIdx.x & 31;
    const int row  = blockIdx.x * 8 + wave;

    const float4 u = ((const float4*)(users + (size_t)row * DIM))[lane];
    const float4 p = ((const float4*)(items + (size_t)row * DIM))[lane];

    float su  = u.x*u.x + u.y*u.y + u.z*u.z + u.w*u.w;
    float sp  = p.x*p.x + p.y*p.y + p.z*p.z + p.w*p.w;
    float sup = u.x*p.x + u.y*p.y + u.z*p.z + u.w*p.w;
    su  = wave_sum(su);
    sp  = wave_sum(sp);
    sup = wave_sum(sup);

    const float inv_u = 1.0f / fmaxf(sqrtf(su), 1e-12f);
    const float inv_p = 1.0f / fmaxf(sqrtf(sp), 1e-12f);

    v4h hu, hp;
    hu[0] = (_Float16)(u.x * inv_u); hu[1] = (_Float16)(u.y * inv_u);
    hu[2] = (_Float16)(u.z * inv_u); hu[3] = (_Float16)(u.w * inv_u);
    hp[0] = (_Float16)(p.x * inv_p); hp[1] = (_Float16)(p.y * inv_p);
    hp[2] = (_Float16)(p.z * inv_p); hp[3] = (_Float16)(p.w * inv_p);
    ((v4h*)(Xu + (size_t)row * DIM))[lane] = hu;
    ((v4h*)(Xi + (size_t)row * DIM))[lane] = hp;

    // ||u_n - p_n||^2 = 2 - 2 * (u_n . p_n)  (unit rows)
    if (lane == 0) atomicAdd(&acc[0], 2.0f - 2.0f * sup * inv_u * inv_p);
}

// Gram + exp + sum: each wave computes a 16x64 tile of X X^T (K=128) with
// V_WMMA_F32_16X16X32_F16, then accumulates sum(exp(4*(dot-1))) over the tile.
__global__ void au_gram_exp(const _Float16* __restrict__ Xu,
                            const _Float16* __restrict__ Xi,
                            float* __restrict__ acc) {
    const _Float16* X    = blockIdx.z ? Xi : Xu;
    float*          accS = blockIdx.z ? (acc + 2) : (acc + 1);

    const int wave = threadIdx.x >> 5;
    const int lane = threadIdx.x & 31;
    const int lr   = lane & 15;
    const int hi   = lane >> 4;

    const int m0 = (blockIdx.y * 8 + wave) * 16;   // row tile
    const int n0 = blockIdx.x * 64;                // 4 col tiles

    const _Float16* arow = X + (size_t)(m0 + lr) * DIM;
    const _Float16* brow = X + (size_t)(n0 + lr) * DIM;

    v8f c0 = {}, c1 = {}, c2 = {}, c3 = {};

#pragma unroll
    for (int k0 = 0; k0 < DIM; k0 += 32) {
        // A fragment (16x32 f16): lane lr row, K = kbA..kbA+15
        const int kbA = k0 + (hi ? 16 : 0);
        v8h alo = *(const v8h*)(arow + kbA);
        v8h ahi = *(const v8h*)(arow + kbA + 8);
        v16h a = __builtin_shufflevector(alo, ahi,
                 0, 1, 2, 3, 4, 5, 6, 7, 8, 9, 10, 11, 12, 13, 14, 15);

        // B fragment (32x16 f16): B = X^T, so lane needs a contiguous row slice
        const int kbB = k0 + (hi ? 16 : 0);
        v16h b0 = *(const v16h*)(brow + (size_t)0  * DIM + kbB);
        v16h b1 = *(const v16h*)(brow + (size_t)16 * DIM + kbB);
        v16h b2 = *(const v16h*)(brow + (size_t)32 * DIM + kbB);
        v16h b3 = *(const v16h*)(brow + (size_t)48 * DIM + kbB);

        c0 = __builtin_amdgcn_wmma_f32_16x16x32_f16(false, a, false, b0, (short)0, c0, false, false);
        c1 = __builtin_amdgcn_wmma_f32_16x16x32_f16(false, a, false, b1, (short)0, c1, false, false);
        c2 = __builtin_amdgcn_wmma_f32_16x16x32_f16(false, a, false, b2, (short)0, c2, false, false);
        c3 = __builtin_amdgcn_wmma_f32_16x16x32_f16(false, a, false, b3, (short)0, c3, false, false);
    }

    // exp(4*(dot - 1)) = exp2( (4*log2 e) * (dot - 1) )
    const float K4 = 5.770780163555854f; // 4 / ln(2)
    float s = 0.0f;
#pragma unroll
    for (int r = 0; r < 8; ++r) {
        s += __builtin_amdgcn_exp2f(K4 * (c0[r] - 1.0f));
        s += __builtin_amdgcn_exp2f(K4 * (c1[r] - 1.0f));
        s += __builtin_amdgcn_exp2f(K4 * (c2[r] - 1.0f));
        s += __builtin_amdgcn_exp2f(K4 * (c3[r] - 1.0f));
    }
    s = wave_sum(s);
    if (lane == 0) atomicAdd(accS, s);
}

__global__ void au_finalize(const float* __restrict__ acc, float* __restrict__ out) {
    if (threadIdx.x == 0 && blockIdx.x == 0) {
        const float ln2 = 0.6931471805599453f;
        const float n   = (float)N_ROWS;
        float align = acc[0] / n;
        // mean over i<j pairs: (S_full - N) / (N*(N-1))
        const float denom = n * (n - 1.0f);
        float mu = fmaxf((acc[1] - n) / denom, 1e-30f);
        float mi = fmaxf((acc[2] - n) / denom, 1e-30f);
        float Uu = __builtin_amdgcn_logf(mu) * ln2;
        float Ui = __builtin_amdgcn_logf(mi) * ln2;
        float uniform = 0.25f * (Uu + Ui);   // 0.5 * (Uu + Ui) / 2
        out[0] = (align + uniform) / (float)N_ROWS;  // / BATCH_SIZE
    }
}

extern "C" void kernel_launch(void* const* d_in, const int* in_sizes, int n_in,
                              void* d_out, int out_size, void* d_ws, size_t ws_size,
                              hipStream_t stream) {
    const float* users = (const float*)d_in[0];
    const float* items = (const float*)d_in[1];
    // d_in[2] (neg_items) is unused, matching the reference.
    float* out = (float*)d_out;

    char* ws = (char*)d_ws;
    float*    acc = (float*)ws;                                   // 4 floats
    _Float16* Xu  = (_Float16*)(ws + 256);                        // 2 MB
    _Float16* Xi  = (_Float16*)(ws + 256 + (size_t)N_ROWS * DIM * sizeof(_Float16));

    au_init_acc<<<1, 32, 0, stream>>>(acc);
    au_normalize<<<N_ROWS / 8, 256, 0, stream>>>(users, items, Xu, Xi, acc);
    // grid: x = 8192/64 col groups, y = 8192/(8 waves *16) row groups, z = {users, items}
    au_gram_exp<<<dim3(N_ROWS / 64, N_ROWS / 128, 2), 256, 0, stream>>>(Xu, Xi, acc);
    au_finalize<<<1, 32, 0, stream>>>(acc, out);
}
